// ManualVanillaRNN_85031762526356
// MI455X (gfx1250) — matrix operand order repaired and checked by reading, not compiled, not run on hardware
//
#include <hip/hip_runtime.h>
#include <hip/hip_bf16.h>
#include <math.h>

// ---------------------------------------------------------------------------
// ManualVanillaRNN on gfx1250 (MI455X)
//   B=256, S=512, H=512, V=128
//   Phase 1: EW[v,n] = embedding[v,:] @ W_ih[:,n] + b_ih[n] + b_hh[n]   (f32)
//   Phase 2: W_hh -> W_hh^T bf16, W_ho -> W_ho^T bf16                   (ws)
//   Phase 3: recurrence, 16 blocks x 8 waves, WMMA bf16->f32.
//     Fused step (single K-loop, A-fragment loaded once, 6 wmma chains):
//        y_t = h_t @ W_ho + b_o          (stored)
//        h_{t+1} = tanh(ew_{t+1} + h_t @ W_hh)
//     - h double-buffered in LDS -> 1 barrier / step
//     - weights streamed from L2 with a 2-iteration load->use distance,
//       enforced by sched_barrier; LICM defeated by an opaque u32 offset
//       (keeps global addrspace -> global_load_b128, not flat_load)
//     - amdgpu_waves_per_eu(2): full register budget, no spills
// ---------------------------------------------------------------------------

typedef __attribute__((ext_vector_type(16))) __bf16 v16bf;
typedef __attribute__((ext_vector_type(8)))  float  v8f;

#define BATCH 256
#define SEQ   512
#define HID   512
#define VOCAB 128
#define HS    520   // LDS row stride (ushorts): 1040B = 65*16B -> conflict-free b128 rows

union BF16x16 {
    v16bf v;
    uint4 q[2];
};

__device__ __forceinline__ unsigned short f2bf(float f) {
    unsigned int u = __float_as_uint(f);
    u += 0x7FFFu + ((u >> 16) & 1u);          // round-to-nearest-even
    return (unsigned short)(u >> 16);
}
__device__ __forceinline__ float bf2f(unsigned short h) {
    return __uint_as_float(((unsigned int)h) << 16);
}

__device__ __forceinline__ float fast_tanh(float v) {
#if __has_builtin(__builtin_amdgcn_tanhf)
    return __builtin_amdgcn_tanhf(v);         // v_tanh_f32 (gfx1250 TRANS)
#else
    const float c = fminf(fmaxf(v, -15.f), 15.f);
    const float e = __expf(2.f * c);          // v_exp_f32
    return 1.f - 2.f / (e + 1.f);
#endif
}

// Loop-variant zero: defeats LICM of loads whose address includes it, while
// keeping the pointer base visible (global addrspace inference intact).
__device__ __forceinline__ unsigned int opaque_zero() {
    unsigned int z = 0;
    asm volatile("" : "+v"(z));
    return z;
}

__device__ __forceinline__ void sched_fence() {
#if __has_builtin(__builtin_amdgcn_sched_barrier)
    __builtin_amdgcn_sched_barrier(0);        // nothing crosses this point
#endif
}

// --------------------------- Phase 1: EW precompute -------------------------
__global__ __launch_bounds__(256) void ew_kernel(const float* __restrict__ emb,
                                                 const float* __restrict__ Wih,
                                                 const float* __restrict__ bih,
                                                 const float* __restrict__ bhh,
                                                 float* __restrict__ EW) {
    int idx = blockIdx.x * blockDim.x + threadIdx.x;   // 128*512 threads
    int v = idx >> 9, n = idx & 511;
    const float* e = emb + (size_t)v * HID;
    float s = bih[n] + bhh[n];
    #pragma unroll 8
    for (int k = 0; k < HID; ++k) s = fmaf(e[k], Wih[(size_t)k * HID + n], s);
    EW[idx] = s;
}

// ------------------ Phase 2: transpose + convert to bf16 --------------------
__global__ __launch_bounds__(256) void conv_kernel(const float* __restrict__ Whh,
                                                   const float* __restrict__ Who,
                                                   unsigned short* __restrict__ WhhT,
                                                   unsigned short* __restrict__ WhoT) {
    int idx = blockIdx.x * blockDim.x + threadIdx.x;   // 327680 threads
    if (idx < HID * HID) {
        int n = idx >> 9, k = idx & 511;
        WhhT[idx] = f2bf(Whh[(size_t)k * HID + n]);
    } else {
        int i2 = idx - HID * HID;
        int v = i2 >> 9, k = i2 & 511;
        WhoT[i2] = f2bf(Who[(size_t)k * VOCAB + v]);
    }
}

// --------------------------- Phase 3: recurrence ----------------------------
__device__ __forceinline__ void gather_ew(const int* __restrict__ x,
                                          const float* __restrict__ EW,
                                          int b0, int hi, int lo, int wave, int t,
                                          v8f nxt[4]) {
    int toks[8];
    #pragma unroll
    for (int j = 0; j < 8; ++j)
        toks[j] = x[(size_t)(b0 + j + hi * 8) * SEQ + t];
    #pragma unroll
    for (int i = 0; i < 4; ++i) {
        const int n = (wave * 4 + i) * 16 + lo;
        #pragma unroll
        for (int j = 0; j < 8; ++j)
            nxt[i][j] = EW[(size_t)toks[j] * HID + n];
    }
}

__global__ __launch_bounds__(256) __attribute__((amdgpu_waves_per_eu(2)))
void rnn_kernel(const int* __restrict__ x,
                const float* __restrict__ EW,
                const unsigned short* __restrict__ WhhT,
                const unsigned short* __restrict__ WhoT,
                const float* __restrict__ b_o,
                float* __restrict__ out,
                float* __restrict__ hT_out) {
    __shared__ unsigned short hbuf[2][16 * HS];   // double-buffered h (bf16)

    const int tid  = threadIdx.x;
    const int wave = tid >> 5;                // 0..7 (wave32)
    const int lane = tid & 31;
    const int lo   = lane & 15;
    const int hi   = lane >> 4;
    const int b0   = blockIdx.x * 16;

    const int   ny = wave * 16 + lo;          // y-projection column (8 waves * 16 = 128)
    const float bo = b_o[ny];

    // ---- prologue: h_0 = tanh(EW[x_0]) (h_{-1} == 0), then prefetch ew_1 ----
    {
        v8f e0[4];
        gather_ew(x, EW, b0, hi, lo, wave, 0, e0);
        #pragma unroll
        for (int i = 0; i < 4; ++i) {
            const int n = (wave * 4 + i) * 16 + lo;
            #pragma unroll
            for (int j = 0; j < 8; ++j)
                hbuf[0][(j + hi * 8) * HS + n] = f2bf(fast_tanh(e0[i][j]));
        }
    }
    v8f nxt[4];
    gather_ew(x, EW, b0, hi, lo, wave, 1, nxt);
    __syncthreads();

    int p = 0;
    for (int t = 0; t < SEQ; ++t) {
        const unsigned short* hold = hbuf[p];        // h_t
        unsigned short*       hnew = hbuf[p ^ 1];    // h_{t+1}

        // loop-variant zero offset: weight loads cannot be hoisted out of t-loop
        const unsigned int z = opaque_zero();
        const unsigned short* Whh_s = WhhT + z;
        const unsigned short* Who_s = WhoT + z;

        // consume prefetched ew_{t+1}; immediately start gather of ew_{t+2}
        v8f acc[4];
        #pragma unroll
        for (int i = 0; i < 4; ++i) acc[i] = nxt[i];
        {
            int tt = t + 2; if (tt > SEQ - 1) tt = SEQ - 1;
            gather_ew(x, EW, b0, hi, lo, wave, tt, nxt);
        }

        // ---- fused K-loop: 1 A-load feeds y (2 chains) + 4 recurrence chains --
        v8f y0, y1;
        #pragma unroll
        for (int j = 0; j < 8; ++j) { y0[j] = bo; y1[j] = 0.f; }

        uint4 aq[2];           // A fragment (LDS), 1 chunk ahead
        aq[0] = *(const uint4*)(hold + lo * HS + hi * 16);
        aq[1] = *(const uint4*)(hold + lo * HS + 8 + hi * 16);

        uint4 bq[2][4][2];     // W_hh fragments, 2 chunks ahead
        uint4 cq[2][2];        // W_ho fragments, 2 chunks ahead
        #pragma unroll
        for (int s = 0; s < 2; ++s) {
            #pragma unroll
            for (int i = 0; i < 4; ++i) {
                const uint4* pw = (const uint4*)(Whh_s +
                    (size_t)((wave * 4 + i) * 16 + lo) * HID + s * 32 + hi * 16);
                bq[s][i][0] = pw[0];
                bq[s][i][1] = pw[1];
            }
            const uint4* pc = (const uint4*)(Who_s + (size_t)ny * HID + s * 32 + hi * 16);
            cq[s][0] = pc[0];
            cq[s][1] = pc[1];
        }

        #pragma unroll
        for (int kc = 0; kc < 16; ++kc) {
            const int cur = kc & 1;

            // -- issue next loads FIRST; sched_barrier pins them above the wmmas
            uint4 nb[4][2], nc[2], na[2];
            if (kc + 2 < 16) {
                #pragma unroll
                for (int i = 0; i < 4; ++i) {
                    const uint4* pw = (const uint4*)(Whh_s +
                        (size_t)((wave * 4 + i) * 16 + lo) * HID + (kc + 2) * 32 + hi * 16);
                    nb[i][0] = pw[0];
                    nb[i][1] = pw[1];
                }
                const uint4* pc = (const uint4*)(Who_s + (size_t)ny * HID + (kc + 2) * 32 + hi * 16);
                nc[0] = pc[0];
                nc[1] = pc[1];
            }
            if (kc < 15) {
                na[0] = *(const uint4*)(hold + lo * HS + (kc + 1) * 32 + hi * 16);
                na[1] = *(const uint4*)(hold + lo * HS + (kc + 1) * 32 + 8 + hi * 16);
            }
            sched_fence();

            // -- 5 wmmas for chunk kc (operands loaded >= 1..2 iterations ago)
            BF16x16 a;
            a.q[0] = aq[0];
            a.q[1] = aq[1];

            BF16x16 wb;
            wb.q[0] = cq[cur][0];
            wb.q[1] = cq[cur][1];
            if (cur == 0)
                y0 = __builtin_amdgcn_wmma_f32_16x16x32_bf16(
                    false, a.v, false, wb.v, (short)0, y0, false, false);
            else
                y1 = __builtin_amdgcn_wmma_f32_16x16x32_bf16(
                    false, a.v, false, wb.v, (short)0, y1, false, false);

            #pragma unroll
            for (int i = 0; i < 4; ++i) {
                BF16x16 bb;
                bb.q[0] = bq[cur][i][0];
                bb.q[1] = bq[cur][i][1];
                acc[i] = __builtin_amdgcn_wmma_f32_16x16x32_bf16(
                    false, a.v, false, bb.v, (short)0, acc[i], false, false);
            }

            // -- rotate pipeline registers (pure renaming after unroll)
            if (kc < 15) {
                aq[0] = na[0];
                aq[1] = na[1];
            }
            if (kc + 2 < 16) {
                #pragma unroll
                for (int i = 0; i < 4; ++i) {
                    bq[cur][i][0] = nb[i][0];
                    bq[cur][i][1] = nb[i][1];
                }
                cq[cur][0] = nc[0];
                cq[cur][1] = nc[1];
            }
        }

        // ---- store y_t ----
        #pragma unroll
        for (int j = 0; j < 8; ++j) {
            const size_t o = ((size_t)(b0 + j + hi * 8) * SEQ + t) * VOCAB + ny;
            out[o] = y0[j] + y1[j];
        }

        // ---- h_{t+1} = tanh(pre-act), write to other buffer ----
        #pragma unroll
        for (int i = 0; i < 4; ++i) {
            const int n = (wave * 4 + i) * 16 + lo;
            #pragma unroll
            for (int j = 0; j < 8; ++j)
                hnew[(j + hi * 8) * HS + n] = f2bf(fast_tanh(acc[i][j]));
        }
        __syncthreads();   // single barrier per step
        p ^= 1;
    }

    // ---- hT = h_511 = input buffer of the last iteration = hbuf[p^1] ----
    for (int i = tid; i < 16 * HID; i += 256) {
        const int m = i >> 9, n = i & 511;
        hT_out[(size_t)(b0 + m) * HID + n] = bf2f(hbuf[p ^ 1][m * HS + n]);
    }
}

// ---------------------------------------------------------------------------
extern "C" void kernel_launch(void* const* d_in, const int* in_sizes, int n_in,
                              void* d_out, int out_size, void* d_ws, size_t ws_size,
                              hipStream_t stream) {
    const int*   x    = (const int*)  d_in[0];
    const float* emb  = (const float*)d_in[1];
    const float* Wih  = (const float*)d_in[2];
    const float* bih  = (const float*)d_in[3];
    const float* Whh  = (const float*)d_in[4];
    const float* bhh  = (const float*)d_in[5];
    const float* Who  = (const float*)d_in[6];
    const float* bo   = (const float*)d_in[7];

    // workspace layout (total < 1 MB)
    float*          EW   = (float*)d_ws;                                     // 256 KB
    unsigned short* WhhT = (unsigned short*)((char*)d_ws + 262144);          // 512 KB
    unsigned short* WhoT = (unsigned short*)((char*)d_ws + 262144 + 524288); // 128 KB

    float* out    = (float*)d_out;
    float* hT_out = out + (size_t)BATCH * SEQ * VOCAB;

    ew_kernel<<<(VOCAB * HID) / 256, 256, 0, stream>>>(emb, Wih, bih, bhh, EW);
    conv_kernel<<<(HID * HID + HID * VOCAB) / 256, 256, 0, stream>>>(Whh, Who, WhhT, WhoT);
    rnn_kernel<<<BATCH / 16, 256, 0, stream>>>(x, EW, WhhT, WhoT, bo, out, hT_out);
}
